// ChunkModule_49237505082087
// MI455X (gfx1250) — compile-verified, matrix operand order checked by reading, not run on hardware
//
#include <hip/hip_runtime.h>
#include <hip/hip_bf16.h>
#include <cstdint>

// ---------------------------------------------------------------------------
// CDNA5 (gfx1250, wave32) implementation of the Mamba-2 chunk module.
// GEMM-shaped work (projections, FFNs, conv1d-as-implicit-GEMM, SSD chunk
// einsums) runs through a bf16 WMMA tile kernel (v_wmma_f32_16x16x32_bf16,
// 128x64 block tile, 8 waves). B operands are pre-packed transposed [N][K]
// bf16 so tiles are contiguous 2-D blocks; they (and the bf16 decay-matrix A
// tiles) are moved global->LDS by the Tensor Data Mover when available.
// ---------------------------------------------------------------------------

typedef __attribute__((ext_vector_type(16))) __bf16 v16bf;
typedef __attribute__((ext_vector_type(8)))  float  v8f;

union Frag { v16bf v; uint4 q[2]; };

#if defined(__gfx1250__) && __has_builtin(__builtin_amdgcn_tensor_load_to_lds) && \
    __has_builtin(__builtin_amdgcn_s_wait_tensorcnt)
#define USE_TDM 1
#else
#define USE_TDM 0
#endif

#if USE_TDM
typedef __attribute__((ext_vector_type(4))) unsigned int v4u_t;
typedef __attribute__((ext_vector_type(8))) int v8i_t;
typedef __attribute__((ext_vector_type(4))) int v4i_t;

// Low 32 bits of a generic pointer to LDS == LDS byte offset on AMDGPU.
__device__ __forceinline__ unsigned lds_byte_off(const void* p) {
  return (unsigned)(unsigned long long)(uintptr_t)p;
}

// 2-D TDM tile load: bf16 elements, tile_k x tile_n tile of a row-major
// [valid_n rows x stride_k] tensor starting at gaddr. Hardware returns zero
// for reads beyond (valid_k, valid_n) -> free tail handling.
__device__ __forceinline__ void tdm_load_2d(unsigned ldsoff, const void* gaddr,
                                            int valid_k, int valid_n,
                                            int stride_k, int tile_k, int tile_n) {
  unsigned long long ga = (unsigned long long)(uintptr_t)gaddr;
  v4u_t g0;
  g0[0] = 1u;                                   // count=1, user mode, load
  g0[1] = ldsoff;                               // lds_addr (bytes)
  g0[2] = (unsigned)(ga & 0xffffffffu);         // global_addr[31:0]
  g0[3] = (unsigned)((ga >> 32) & 0x01ffffffu)  // global_addr[56:32]
          | (2u << 30);                         // type = 2 ("image")
  v8i_t g1;
  g1[0] = (int)(1u << 16);                      // data_size = 2 bytes
  g1[1] = (int)(((unsigned)valid_k & 0xffffu) << 16);            // td0[15:0]
  g1[2] = (int)((((unsigned)valid_k >> 16) & 0xffffu)            // td0[31:16]
          | (((unsigned)valid_n & 0xffffu) << 16));              // td1[15:0]
  g1[3] = (int)((((unsigned)valid_n >> 16) & 0xffffu)            // td1[31:16]
          | (((unsigned)tile_k & 0xffffu) << 16));               // tile_dim0
  g1[4] = (int)((unsigned)tile_n & 0xffffu);                     // tile_dim1
  g1[5] = stride_k;                             // tensor_dim0_stride[31:0]
  g1[6] = 0;
  g1[7] = 0;
  v4i_t z4 = {0, 0, 0, 0};
  v8i_t z8 = {0, 0, 0, 0, 0, 0, 0, 0};
  // 6-arg form (clang-23 / therock headers): (g0, g1, g2, g3, g4, cpol)
  __builtin_amdgcn_tensor_load_to_lds(g0, g1, z4, z4, z8, 0);
}
#endif

__device__ __forceinline__ float gelu_f(float x) {
  return 0.5f * x * (1.f + erff(x * 0.70710678118654752f));
}
__device__ __forceinline__ float silu_f(float x) {
  return x / (1.f + expf(-x));
}

// EPI bits: 1=bias, 2=gelu, 4=residual-add.  MODE: 0=dense f32 A, 1=implicit
// conv1d A (L=1024, D=512 hardcoded shifts), 2=A already bf16 (decay matrix).
// W is transposed bf16 [N][K], ldb = K-stride.
template<int EPI, int MODE>
__global__ __launch_bounds__(256, 2)
void k_gemm(const void* __restrict__ Ap, const __bf16* __restrict__ Wp,
            float* __restrict__ Cp,
            int M, int N, int K, int lda, int ldb, int ldc,
            long long sA, long long sB, long long sC,
            const float* __restrict__ bias,
            const float* __restrict__ res, int ldres,
            int convPad, int hdiv, long long sAh) {
  __shared__ __align__(16) __bf16 shA[128 * 32];
  __shared__ __align__(16) __bf16 shB[64 * 32];

  const int tid = threadIdx.x;
  const int n0 = blockIdx.x * 64;
  const int m0 = blockIdx.y * 128;
  const int z  = blockIdx.z;

  long long aoff = (long long)z * sA;
  if (MODE == 2 && hdiv > 0) aoff += (long long)(n0 / hdiv) * sAh;
  const float*  Af  = (const float*)Ap + aoff;
  const __bf16* Abf = (const __bf16*)Ap + aoff;
  const __bf16* Wb  = Wp + (long long)z * sB;
  float*        Cm  = Cp + (long long)z * sC;

  const int w = tid >> 5, lane = tid & 31, hs = lane >> 4, l16 = lane & 15;

  // Hoisted per-thread A-loader coordinates (k-invariant).
  const int ar  = tid >> 3;        // 0..31 row group
  const int ac4 = (tid & 7) * 4;   // k sub-offset
  int mrow[4], trow[4], brow[4];
#pragma unroll
  for (int it = 0; it < 4; ++it) {
    int m = m0 + ar + it * 32;
    mrow[it] = m;
    if (MODE == 1) { brow[it] = m >> 10; trow[it] = m & 1023; }  // L = 1024
  }

  v8f acc[4] = {};

  for (int k0 = 0; k0 < K; k0 += 32) {
#if USE_TDM
    if (w == 0) {
      if (MODE == 2)
        tdm_load_2d(lds_byte_off(shA), Abf + (long long)m0 * lda + k0,
                    K - k0, M - m0, lda, 32, 128);
      tdm_load_2d(lds_byte_off(shB), Wb + (long long)n0 * ldb + k0,
                  K - k0, N - n0, ldb, 32, 64);
    }
#endif
    // ---- stage A tile (128 x 32) f32 -> bf16 (MODE 0/1) -------------------
    if (MODE != 2) {
#pragma unroll
      for (int it = 0; it < 4; ++it) {
        int gk = k0 + ac4;
        float4 v = make_float4(0.f, 0.f, 0.f, 0.f);
        if (MODE == 1) {
          int dk = gk >> 9, ii = gk & 511;                       // D = 512
          int ts = trow[it] + dk - convPad;
          if (mrow[it] < M && ts >= 0 && ts < 1024)
            v = *(const float4*)(Af + (((long long)brow[it] << 10) + ts) * 512 + ii);
        } else {
          if (mrow[it] < M && gk < K)
            v = *(const float4*)(Af + (long long)mrow[it] * lda + gk);
        }
        union { __bf16 h[4]; uint2 u; } cv;
        cv.h[0] = (__bf16)v.x; cv.h[1] = (__bf16)v.y;
        cv.h[2] = (__bf16)v.z; cv.h[3] = (__bf16)v.w;
        *(uint2*)&shA[(ar + it * 32) * 32 + ac4] = cv.u;
      }
      if (MODE == 0 && k0 + 32 < K)
        __builtin_prefetch(Af + (long long)mrow[0] * lda + k0 + 32, 0, 1);
    }
#if !USE_TDM
    if (MODE == 2) {
      int r = tid >> 1, h0 = (tid & 1) * 16;
      int m = m0 + r;
      uint4 v0 = make_uint4(0, 0, 0, 0), v1 = v0;
      if (m < M) {
        const uint4* src = (const uint4*)(Abf + (long long)m * lda + k0 + h0);
        v0 = src[0]; v1 = src[1];
      }
      *(uint4*)&shA[r * 32 + h0]     = v0;
      *(uint4*)&shA[r * 32 + h0 + 8] = v1;
    }
    // ---- stage B tile (64 x 32) from transposed bf16 [N][K] ---------------
    {
      int nn = tid >> 2, k8 = (tid & 3) * 8;
      int n = n0 + nn, gk = k0 + k8;
      uint4 v = make_uint4(0, 0, 0, 0);
      if (n < N && gk < K) v = *(const uint4*)(Wb + (long long)n * ldb + gk);
      *(uint4*)&shB[nn * 32 + k8] = v;
    }
#endif
#if USE_TDM
    if (w == 0) __builtin_amdgcn_s_wait_tensorcnt(0);
#endif
    __syncthreads();

    // ---- fragments per ISA layout; 4 WMMAs per wave per k-step ------------
    Frag fa;
    fa.q[0] = *(const uint4*)&shA[(w * 16 + l16) * 32 + hs * 8];
    fa.q[1] = *(const uint4*)&shA[(w * 16 + l16) * 32 + 16 + hs * 8];
#pragma unroll
    for (int nt = 0; nt < 4; ++nt) {
      Frag fb;
      fb.q[0] = *(const uint4*)&shB[(nt * 16 + l16) * 32 + hs * 8];
      fb.q[1] = *(const uint4*)&shB[(nt * 16 + l16) * 32 + 16 + hs * 8];
      acc[nt] = __builtin_amdgcn_wmma_f32_16x16x32_bf16(
          false, fa.v, false, fb.v, (short)0, acc[nt], false, false);
    }
    __syncthreads();
  }

  // ---- epilogue: C/D layout VGPR r -> M = r + 8*(lane>=16) ----------------
#pragma unroll
  for (int nt = 0; nt < 4; ++nt) {
    int n = n0 + nt * 16 + l16;
    if (n >= N) continue;
#pragma unroll
    for (int r = 0; r < 8; ++r) {
      int m = m0 + w * 16 + r + 8 * hs;
      if (m >= M) continue;
      float f = acc[nt][r];
      if (EPI & 1) f += bias[n];
      if (EPI & 2) f = gelu_f(f);
      if (EPI & 4) f += res[(long long)m * ldres + n];
      Cm[(long long)m * ldc + n] = f;
    }
  }
}

// ---------------------------------------------------------------------------
// Elementwise / reduction kernels
// ---------------------------------------------------------------------------

__global__ void k_ln(const float* __restrict__ x, const float* __restrict__ w,
                     const float* __restrict__ b, float* __restrict__ y, int cols) {
  __shared__ float red[256];
  __shared__ float s_m, s_r;
  const float* xr = x + (long long)blockIdx.x * cols;
  float s = 0.f;
  for (int c = threadIdx.x; c < cols; c += 256) s += xr[c];
  red[threadIdx.x] = s; __syncthreads();
  for (int o = 128; o > 0; o >>= 1) { if (threadIdx.x < o) red[threadIdx.x] += red[threadIdx.x + o]; __syncthreads(); }
  if (threadIdx.x == 0) s_m = red[0] / cols;
  __syncthreads();
  float m = s_m; s = 0.f;
  for (int c = threadIdx.x; c < cols; c += 256) { float d = xr[c] - m; s += d * d; }
  red[threadIdx.x] = s; __syncthreads();
  for (int o = 128; o > 0; o >>= 1) { if (threadIdx.x < o) red[threadIdx.x] += red[threadIdx.x + o]; __syncthreads(); }
  if (threadIdx.x == 0) s_r = rsqrtf(red[0] / cols + 1e-5f);
  __syncthreads();
  float rs = s_r;
  float* yr = y + (long long)blockIdx.x * cols;
  for (int c = threadIdx.x; c < cols; c += 256) yr[c] = (xr[c] - m) * rs * w[c] + b[c];
}

__global__ void k_bnorm(const float* __restrict__ x, const float* __restrict__ g,
                        const float* __restrict__ b, float* __restrict__ y,
                        int rows, int D, int do_gelu) {
  int d = blockIdx.x;
  __shared__ float r1[256], r2[256];
  __shared__ float sm, sr;
  float s = 0.f, s2 = 0.f;
  for (int r = threadIdx.x; r < rows; r += 256) {
    float v = x[(long long)r * D + d]; s += v; s2 += v * v;
  }
  r1[threadIdx.x] = s; r2[threadIdx.x] = s2; __syncthreads();
  for (int o = 128; o > 0; o >>= 1) {
    if (threadIdx.x < o) { r1[threadIdx.x] += r1[threadIdx.x + o]; r2[threadIdx.x] += r2[threadIdx.x + o]; }
    __syncthreads();
  }
  if (threadIdx.x == 0) {
    float m = r1[0] / rows, var = r2[0] / rows - m * m;
    sm = m; sr = rsqrtf(var + 1e-5f);
  }
  __syncthreads();
  float m = sm, rs = sr, gg = g[d], bb = b[d];
  for (int r = threadIdx.x; r < rows; r += 256) {
    float v = (x[(long long)r * D + d] - m) * rs * gg + bb;
    if (do_gelu) v = gelu_f(v);
    y[(long long)r * D + d] = v;
  }
}

__global__ void k_add(const float* a, const float* b, float* o, long long n) {
  long long i = (long long)blockIdx.x * 256 + threadIdx.x;
  if (i < n) o[i] = a[i] + b[i];
}

__global__ void k_reverse(const float* in, float* out, int Bb, int Lm, int cols,
                          int old, int coff) {
  long long i = (long long)blockIdx.x * 256 + threadIdx.x;
  long long tot = (long long)Bb * Lm * cols;
  if (i >= tot) return;
  int c = (int)(i % cols);
  long long r = i / cols;
  int t = (int)(r % Lm), b = (int)(r / Lm);
  out[((long long)b * Lm + (Lm - 1 - t)) * old + coff + c] = in[i];
}

__global__ void k_copy_cols(const float* in, float* out, long long rows, int cols,
                            int old, int coff) {
  long long i = (long long)blockIdx.x * 256 + threadIdx.x;
  long long tot = rows * cols;
  if (i >= tot) return;
  long long r = i / cols; int c = (int)(i % cols);
  out[r * old + coff + c] = in[i];
}

__global__ void k_copy_seq(const float* src, float* dst, int Bb, int Ls, int cols,
                           long long srcBS, long long dstBS) {
  long long i = (long long)blockIdx.x * 256 + threadIdx.x;
  long long tot = (long long)Bb * Ls * cols;
  if (i >= tot) return;
  int c = (int)(i % cols);
  long long r = i / cols;
  int t = (int)(r % Ls), b = (int)(r / Ls);
  dst[(long long)b * dstBS + (long long)t * cols + c] =
      src[(long long)b * srcBS + (long long)t * cols + c];
}

__global__ void k_slice_add(const float* xsf, int Ltot, int off, const float* lnl,
                            float* out, int Bb, int Ll, int cols) {
  long long i = (long long)blockIdx.x * 256 + threadIdx.x;
  long long tot = (long long)Bb * Ll * cols;
  if (i >= tot) return;
  int c = (int)(i % cols);
  long long r = i / cols;
  int t = (int)(r % Ll), b = (int)(r / Ll);
  out[i] = xsf[((long long)b * Ltot + off + t) * cols + c] + lnl[i];
}

__global__ void k_outer(const float* xe, const float* w, float* out, int Bb, int E, int D) {
  long long i = (long long)blockIdx.x * 256 + threadIdx.x;
  long long tot = (long long)Bb * E * D;
  if (i >= tot) return;
  int d = (int)(i % D);
  long long r = i / D;
  out[i] = xe[r] * w[d];
}

// dense weight f32 [K][N] -> transposed bf16 [N][K]
__global__ void k_packWT(const float* in, __bf16* out, int K, int N) {
  long long i = (long long)blockIdx.x * 256 + threadIdx.x;
  long long tot = (long long)N * K;
  if (i >= tot) return;
  int n = (int)(i / K), k = (int)(i % K);
  out[i] = (__bf16)in[(long long)k * N + n];
}

// conv weight (O,I,ks) f32 -> transposed bf16 [O][dk*I + i]  (ldb = I*ks)
__global__ void k_pack_convw(const float* w, __bf16* out, int O, int I, int ks) {
  long long i = (long long)blockIdx.x * 256 + threadIdx.x;
  long long tot = (long long)O * I * ks;
  if (i >= tot) return;
  int o = (int)(i / ((long long)I * ks));
  long long rem = i - (long long)o * I * ks;
  int dk = (int)(rem / I), ii = (int)(rem % I);
  out[i] = (__bf16)w[((long long)o * I + ii) * ks + dk];
}

// B-matrix slice of xBC -> bf16 [s][n] per (b,c) batch (already W^T layout)
__global__ void k_packBm(const float* xbc, int cd, __bf16* out, int ds, int nbc) {
  long long i = (long long)blockIdx.x * 256 + threadIdx.x;
  long long per = 256LL * ds;
  long long tot = per * nbc;
  if (i >= tot) return;
  int z = (int)(i / per);
  long long rem = i - (long long)z * per;
  int s = (int)(rem / ds), n = (int)(rem % ds);
  out[i] = (__bf16)xbc[((long long)z * 256 + s) * cd + 1024 + n];
}

// xdt f32 [rows][1024] -> bf16 [bc][j][s] (transposed per 256-chunk)
__global__ void k_packXdtT(const float* xdt, __bf16* out, int nbc) {
  long long i = (long long)blockIdx.x * 256 + threadIdx.x;
  long long per = 1024LL * 256;
  long long tot = per * nbc;
  if (i >= tot) return;
  int z = (int)(i / per);
  long long rem = i - (long long)z * per;
  int j = (int)(rem / 256), s = (int)(rem % 256);
  out[i] = (__bf16)xdt[((long long)z * 256 + s) * 1024 + j];
}

// causal depthwise conv (k=4) + bias + silu on strided xBC slice of zxbcdt
__global__ void k_dwconv(const float* zx, int din, int zoff, const float* w,
                         const float* b, float* out, int Bb, int Lm, int cd) {
  long long i = (long long)blockIdx.x * 256 + threadIdx.x;
  long long tot = (long long)Bb * Lm * cd;
  if (i >= tot) return;
  int c = (int)(i % cd);
  long long r = i / cd;
  int t = (int)(r % Lm), bb = (int)(r / Lm);
  float acc = b[c];
#pragma unroll
  for (int j = 0; j < 4; ++j) {
    int ts = t - 3 + j;
    if (ts >= 0) acc += zx[((long long)bb * Lm + ts) * din + zoff + c] * w[c * 4 + j];
  }
  out[i] = silu_f(acc);
}

// dt = softplus(dt_raw + bias); xdt = xh*dt; dA = -exp(A_log)*dt
__global__ void k_dt_prep(const float* zx, int din, int ds, int nh, int hd,
                          const float* xbc, int cd, const float* dtb,
                          const float* Alog, float* xdt, float* dA, long long rows) {
  long long i = (long long)blockIdx.x * 256 + threadIdx.x;
  long long tot = rows * 1024;
  if (i >= tot) return;
  long long r = i / 1024;
  int j = (int)(i % 1024);
  int h = j / hd;
  float dtr = zx[r * din + 2048 + 2 * ds + h] + dtb[h];
  float dtv = (dtr > 20.f) ? dtr : log1pf(expf(dtr));
  xdt[i] = xbc[r * cd + j] * dtv;
  if ((j % hd) == 0) dA[r * nh + h] = -expf(Alog[h]) * dtv;
}

// per (b,h,c): inclusive scan of dA over 256-chunk
__global__ void k_cumsum(const float* dA, int nh, int Lm, int nc, float* Acs) {
  int bhc = blockIdx.x;
  int b = bhc / (nh * nc);
  int rem = bhc % (nh * nc);
  int h = rem / nc, c = rem % nc;
  __shared__ float s[256];
  int t = threadIdx.x;
  s[t] = dA[((long long)b * Lm + c * 256 + t) * nh + h];
  __syncthreads();
  for (int o = 1; o < 256; o <<= 1) {
    float v = (t >= o) ? s[t - o] : 0.f;
    __syncthreads();
    s[t] += v;
    __syncthreads();
  }
  Acs[(long long)bhc * 256 + t] = s[t];
}

// Mh[(bc*nh+h)][l][s] = (s<=l) ? G[bc][l][s]*exp(Acs[l]-Acs[s]) : 0   (bf16)
__global__ void k_decay_mask(const float* G, const float* Acs, __bf16* Mh,
                             int nh, int nc) {
  int l = blockIdx.x, s = threadIdx.x;
  int bch = blockIdx.y;
  int bc = bch / nh, h = bch % nh;
  int b = bc / nc, c = bc % nc;
  const float* A = Acs + ((long long)((b * nh + h) * nc + c)) * 256;
  float v = 0.f;
  if (s <= l) v = G[(long long)bc * 65536 + l * 256 + s] * expf(A[l] - A[s]);
  Mh[(long long)bch * 65536 + (long long)l * 256 + s] = (__bf16)v;
}

// states[bc][h][p][n] = sum_s B[s,n] * exp(Acs_last - Acs[s]) * xdt[s, h*hd+p]
__global__ void k_states(const float* xbc, int cd, const float* xdt,
                         const float* Acs, float* st, int ds, int hd, int nh, int nc) {
  int bc = blockIdx.x, h = blockIdx.y;
  int b = bc / nc, c = bc % nc;
  const float* A = Acs + ((long long)((b * nh + h) * nc + c)) * 256;
  __shared__ float dec[256];
  float Al = A[255];
  dec[threadIdx.x] = expf(Al - A[threadIdx.x]);
  __syncthreads();
  int tot = hd * ds;
  for (int o = threadIdx.x; o < tot; o += 256) {
    int p = o / ds, n = o % ds;
    float acc = 0.f;
    for (int s = 0; s < 256; ++s) {
      acc += xbc[((long long)bc * 256 + s) * cd + 1024 + n] * dec[s] *
             xdt[((long long)bc * 256 + s) * 1024 + h * hd + p];
    }
    st[((long long)bc * nh + h) * tot + o] = acc;
  }
}

// inter-chunk recurrence: init[z] = sum_{k=1..z} exp(Q[z]-Q[k]) * states[k-1]
__global__ void k_chain(const float* Acs, const float* st, float* ist,
                        int ds, int hd, int nh, int nc) {
  int bh = blockIdx.x;
  int b = bh / nh, h = bh % nh;
  __shared__ float Q[32];
  if (threadIdx.x == 0) {
    Q[0] = 0.f;
    for (int z = 1; z <= nc; ++z)
      Q[z] = Q[z - 1] + Acs[((long long)((b * nh + h) * nc + z - 1)) * 256 + 255];
  }
  __syncthreads();
  int tot = hd * ds;
  for (int o = threadIdx.x; o < tot; o += 256) {
    for (int z = 0; z < nc; ++z) {
      float acc = 0.f;
      for (int k = 1; k <= z; ++k)
        acc += expf(Q[z] - Q[k]) *
               st[(((long long)b * nc + k - 1) * nh + h) * tot + o];
      ist[(((long long)b * nc + z) * nh + h) * tot + o] = acc;
    }
  }
}

// Y = Ydiag + C@init^T * exp(Acs) + D*xh; gate with silu(z); RMS-norm * norm_w
__global__ void k_ssd_finish(const float* Yd, const float* xbc, int cd,
                             const float* zx, int din, const float* Acs,
                             const float* ist, const float* Dw,
                             const float* nw, int ds, int hd, int nh, int nc,
                             int Lm, float* out) {
  long long r = blockIdx.x;
  int b = (int)(r / Lm), t = (int)(r % Lm);
  int c = t / 256, l = t % 256;
  __shared__ float Csh[64];
  __shared__ float red[256];
  __shared__ float srs;
  if (threadIdx.x < ds) Csh[threadIdx.x] = xbc[r * cd + 1024 + ds + threadIdx.x];
  __syncthreads();
  float gv[4];
  float ss = 0.f;
#pragma unroll
  for (int q = 0; q < 4; ++q) {
    int j = q * 256 + threadIdx.x;
    int h = j / hd, p = j % hd;
    float od = expf(Acs[((long long)((b * nh + h) * nc + c)) * 256 + l]);
    const float* is = ist + ((((long long)b * nc + c) * nh + h) * hd + p) * ds;
    float yo = 0.f;
    for (int n = 0; n < ds; ++n) yo += Csh[n] * is[n];
    float xh = xbc[r * cd + j];
    float zz = zx[r * din + j];
    float tv = Yd[r * 1024 + j] + yo * od + Dw[h] * xh;
    float g = tv * silu_f(zz);
    gv[q] = g;
    ss += g * g;
  }
  red[threadIdx.x] = ss; __syncthreads();
  for (int o = 128; o > 0; o >>= 1) { if (threadIdx.x < o) red[threadIdx.x] += red[threadIdx.x + o]; __syncthreads(); }
  if (threadIdx.x == 0) srs = rsqrtf(red[0] / 1024.f + 1e-5f);
  __syncthreads();
  float rs = srs;
#pragma unroll
  for (int q = 0; q < 4; ++q) {
    int j = q * 256 + threadIdx.x;
    out[r * 1024 + j] = gv[q] * rs * nw[j];
  }
}

// ---------------------------------------------------------------------------
// Host-side orchestration
// ---------------------------------------------------------------------------

struct Bump {
  char* p; size_t used, cap;
  void* alloc(size_t bytes) {
    size_t a = (used + 255) & ~(size_t)255;
    used = a + bytes;
    return (void*)(p + a);
  }
};

static inline unsigned gblk(long long n) { return (unsigned)((n + 255) / 256); }

static void gemm(hipStream_t st, const void* A, const __bf16* W, float* C,
                 int M, int N, int K, int lda, int ldb, int ldc,
                 long long sA, long long sB, long long sC, int nb,
                 const float* bias, const float* res, int ldres,
                 int epi, int mode, int convPad, int hdiv, long long sAh) {
  dim3 g((N + 63) / 64, (M + 127) / 128, nb), b(256);
#define GCALL(E, Md) k_gemm<E, Md><<<g, b, 0, st>>>(A, W, C, M, N, K, lda, ldb, ldc, \
    sA, sB, sC, bias, res, ldres, convPad, hdiv, sAh)
  if (mode == 1)       { GCALL(3, 1); }
  else if (mode == 2)  { GCALL(0, 2); }
  else {
    switch (epi) {
      case 0: GCALL(0, 0); break;
      case 3: GCALL(3, 0); break;
      case 4: GCALL(4, 0); break;
      case 5: GCALL(5, 0); break;
      default: GCALL(0, 0); break;
    }
  }
#undef GCALL
}

// pack dense f32 [K][N] -> bf16 [N][K]
static void packWT(hipStream_t st, const float* in, __bf16* out, int K, int N) {
  k_packWT<<<gblk((long long)N * K), 256, 0, st>>>(in, out, K, N);
}

struct MambaW { const float *W_in, *conv_w, *conv_b, *dt_bias, *A_log, *D, *norm_w, *W_out; };
struct MambaDims { int ds, hd, nh; };

static void run_mamba2(hipStream_t st, Bump& ws, const float* xin, int Bb, int Lm,
                       const MambaW& mw, MambaDims dm, float* out, int out_ld,
                       const float* res, int ldres) {
  size_t mark = ws.used;
  const int DI = 1024, ds = dm.ds, hd = dm.hd, nh = dm.nh;
  const int cd = DI + 2 * ds, din = 2 * DI + 2 * ds + nh;
  const int nc = Lm / 256, nbc = Bb * nc;
  const long long rows = (long long)Bb * Lm;

  __bf16* Wb = (__bf16*)ws.alloc((size_t)512 * din * 2);
  packWT(st, mw.W_in, Wb, 512, din);
  float* zx = (float*)ws.alloc((size_t)rows * din * 4);
  gemm(st, xin, Wb, zx, (int)rows, din, 512, 512, 512, din, 0, 0, 0, 1,
       nullptr, nullptr, 0, 0, 0, 0, 0, 0);

  float* xbc = (float*)ws.alloc((size_t)rows * cd * 4);
  k_dwconv<<<gblk(rows * cd), 256, 0, st>>>(zx, din, DI, mw.conv_w, mw.conv_b,
                                            xbc, Bb, Lm, cd);

  float* xdt = (float*)ws.alloc((size_t)rows * DI * 4);
  float* dA  = (float*)ws.alloc((size_t)rows * nh * 4);
  k_dt_prep<<<gblk(rows * DI), 256, 0, st>>>(zx, din, ds, nh, hd, xbc, cd,
                                             mw.dt_bias, mw.A_log, xdt, dA, rows);

  float* Acs = (float*)ws.alloc((size_t)Bb * nh * nc * 256 * 4);
  k_cumsum<<<Bb * nh * nc, 256, 0, st>>>(dA, nh, Lm, nc, Acs);

  __bf16* Btb = (__bf16*)ws.alloc((size_t)nbc * ds * 256 * 2);
  k_packBm<<<gblk((long long)nbc * 256 * ds), 256, 0, st>>>(xbc, cd, Btb, ds, nbc);

  float* G = (float*)ws.alloc((size_t)nbc * 65536 * 4);
  gemm(st, xbc + (DI + ds), Btb, G, 256, 256, ds, cd, ds, 256,
       (long long)256 * cd, (long long)256 * ds, 65536LL, nbc,
       nullptr, nullptr, 0, 0, 0, 0, 0, 0);

  __bf16* Mh = (__bf16*)ws.alloc((size_t)nbc * nh * 65536 * 2);
  k_decay_mask<<<dim3(256, nbc * nh), 256, 0, st>>>(G, Acs, Mh, nh, nc);

  __bf16* xdtT = (__bf16*)ws.alloc((size_t)nbc * 1024 * 256 * 2);
  k_packXdtT<<<gblk((long long)nbc * 1024 * 256), 256, 0, st>>>(xdt, xdtT, nbc);

  float* Yd = (float*)ws.alloc((size_t)rows * DI * 4);
  gemm(st, Mh, xdtT, Yd, 256, DI, 256, 256, 256, DI,
       (long long)nh * 65536, 262144LL, (long long)256 * DI, nbc,
       nullptr, nullptr, 0, 0, 2, 0, hd, 65536LL);

  float* Sts = (float*)ws.alloc((size_t)nbc * nh * hd * ds * 4);
  k_states<<<dim3(nbc, nh), 256, 0, st>>>(xbc, cd, xdt, Acs, Sts, ds, hd, nh, nc);

  float* Ist = (float*)ws.alloc((size_t)nbc * nh * hd * ds * 4);
  k_chain<<<Bb * nh, 256, 0, st>>>(Acs, Sts, Ist, ds, hd, nh, nc);

  float* yn = (float*)ws.alloc((size_t)rows * DI * 4);
  k_ssd_finish<<<(unsigned)rows, 256, 0, st>>>(Yd, xbc, cd, zx, din, Acs, Ist,
                                               mw.D, mw.norm_w, ds, hd, nh, nc, Lm, yn);

  __bf16* Wob = (__bf16*)ws.alloc((size_t)DI * 512 * 2);
  packWT(st, mw.W_out, Wob, DI, 512);
  gemm(st, yn, Wob, out, (int)rows, 512, DI, DI, DI, out_ld, 0, 0, 0, 1,
       nullptr, res, ldres, res ? 4 : 0, 0, 0, 0, 0);

  ws.used = mark;
}

static void mamba2_block(hipStream_t st, Bump& ws, const float* x, int Bb, int Lm,
                         const float* lnw, const float* lnb, const MambaW& mw,
                         MambaDims dm, float* out) {
  size_t mark = ws.used;
  long long rows = (long long)Bb * Lm;
  float* xn = (float*)ws.alloc((size_t)rows * 512 * 4);
  k_ln<<<(unsigned)rows, 256, 0, st>>>(x, lnw, lnb, xn, 512);
  run_mamba2(st, ws, xn, Bb, Lm, mw, dm, out, 512, x, 512);
  ws.used = mark;
}

struct ConvW { const float* w[7]; const float* b[7]; const float *g1, *b1, *g2, *b2; };
static const int KS[7] = {3, 5, 7, 7, 15, 3, 1};

static void conv_apply(hipStream_t st, Bump& ws, const float* x, float* y,
                       const float* w, const float* b, int ks, long long rows) {
  size_t mark = ws.used;
  __bf16* wp = (__bf16*)ws.alloc((size_t)512 * 512 * ks * 2);
  k_pack_convw<<<gblk((long long)512 * 512 * ks), 256, 0, st>>>(w, wp, 512, 512, ks);
  gemm(st, x, wp, y, (int)rows, 512, 512 * ks, 512, 512 * ks, 512, 0, 0, 0, 1,
       b, nullptr, 0, 3, 1, (ks - 1) / 2, 0, 0);
  ws.used = mark;
}

static void run_conv_block(hipStream_t st, Bump& ws, const float* x, float* out,
                           const ConvW& c, long long rows) {
  size_t mark = ws.used;
  size_t sz = (size_t)rows * 512;
  float* t1 = (float*)ws.alloc(sz * 4);
  float* t2 = (float*)ws.alloc(sz * 4);
  float* t3 = (float*)ws.alloc(sz * 4);
  float* t4 = (float*)ws.alloc(sz * 4);
  conv_apply(st, ws, x,  t1, c.w[0], c.b[0], KS[0], rows);   // c1
  conv_apply(st, ws, t1, t2, c.w[1], c.b[1], KS[1], rows);   // c2
  conv_apply(st, ws, t2, t3, c.w[2], c.b[2], KS[2], rows);   // c3
  conv_apply(st, ws, t1, t4, c.w[3], c.b[3], KS[3], rows);   // cl1
  conv_apply(st, ws, t4, t2, c.w[4], c.b[4], KS[4], rows);   // cl2
  k_add<<<gblk((long long)sz), 256, 0, st>>>(t3, t2, t1, (long long)sz);
  conv_apply(st, ws, t1, t4, c.w[5], c.b[5], KS[5], rows);   // cf
  k_bnorm<<<512, 256, 0, st>>>(t4, c.g1, c.b1, t1, (int)rows, 512, 0);
  conv_apply(st, ws, t1, t3, c.w[6], c.b[6], KS[6], rows);   // cr
  k_bnorm<<<512, 256, 0, st>>>(t3, c.g2, c.b2, out, (int)rows, 512, 1);
  ws.used = mark;
}

struct BimW {
  const float *l1w, *l1b, *l2w, *l2b;
  MambaW fw, bw;
  const float *f1w, *f1b, *f2w, *f2b;
};

static void run_bimamba(hipStream_t st, Bump& ws, const float* x, float* out,
                        const BimW& bw) {
  size_t mark = ws.used;
  const int Bb = 2, Lm = 1024;
  const long long rows = (long long)Bb * Lm;
  MambaDims dm{16, 64, 16};
  float* xn = (float*)ws.alloc((size_t)rows * 512 * 4);
  k_ln<<<(unsigned)rows, 256, 0, st>>>(x, bw.l1w, bw.l1b, xn, 512);
  float* bi = (float*)ws.alloc((size_t)rows * 1024 * 4);
  run_mamba2(st, ws, xn, Bb, Lm, bw.fw, dm, bi, 1024, nullptr, 0);
  float* xr = (float*)ws.alloc((size_t)rows * 512 * 4);
  k_reverse<<<gblk(rows * 512), 256, 0, st>>>(xn, xr, Bb, Lm, 512, 512, 0);
  float* tb = (float*)ws.alloc((size_t)rows * 512 * 4);
  run_mamba2(st, ws, xr, Bb, Lm, bw.bw, dm, tb, 512, nullptr, 0);
  k_reverse<<<gblk(rows * 512), 256, 0, st>>>(tb, bi, Bb, Lm, 512, 1024, 512);
  float* h1 = (float*)ws.alloc((size_t)rows * 2048 * 4);
  __bf16* w1 = (__bf16*)ws.alloc((size_t)1024 * 2048 * 2);
  packWT(st, bw.f1w, w1, 1024, 2048);
  gemm(st, bi, w1, h1, (int)rows, 2048, 1024, 1024, 1024, 2048, 0, 0, 0, 1,
       bw.f1b, nullptr, 0, 3, 0, 0, 0, 0);
  __bf16* w2 = (__bf16*)ws.alloc((size_t)2048 * 512 * 2);
  packWT(st, bw.f2w, w2, 2048, 512);
  gemm(st, h1, w2, tb, (int)rows, 512, 2048, 2048, 2048, 512, 0, 0, 0, 1,
       bw.f2b, nullptr, 0, 3, 0, 0, 0, 0);
  k_add<<<gblk(rows * 512), 256, 0, st>>>(x, tb, xr, rows * 512);
  k_ln<<<(unsigned)rows, 256, 0, st>>>(xr, bw.l2w, bw.l2b, out, 512);
  ws.used = mark;
}

struct StackW {
  const float *l1w, *l1b, *l2w, *l2b, *l3w, *l3b, *ew, *elw, *elb;
  const float *eblk_lnw, *eblk_lnb; MambaW eblk;
  const float *sblk_lnw, *sblk_lnb; MambaW sblk;
  const float *f1w, *f1b, *f2w, *f2b;
};

static void run_stack(hipStream_t st, Bump& ws, const float* local,
                      const float* glob, const float* xe, float* out,
                      const StackW& sw) {
  size_t mark = ws.used;
  MambaDims dm{64, 128, 8};
  const long long rowsL = 2048;
  float* lnl = (float*)ws.alloc((size_t)rowsL * 512 * 4);
  k_ln<<<(unsigned)rowsL, 256, 0, st>>>(local, sw.l1w, sw.l1b, lnl, 512);
  float* lng = (float*)ws.alloc((size_t)rowsL * 512 * 4);
  k_ln<<<(unsigned)rowsL, 256, 0, st>>>(glob, sw.l2w, sw.l2b, lng, 512);
  float* h = (float*)ws.alloc((size_t)1024 * 512 * 4);
  k_outer<<<gblk(1024LL * 512), 256, 0, st>>>(xe, sw.ew, h, 2, 512, 512);
  float* h2 = (float*)ws.alloc((size_t)1024 * 512 * 4);
  k_ln<<<1024, 256, 0, st>>>(h, sw.elw, sw.elb, h2, 512);
  float* etok = (float*)ws.alloc((size_t)1024 * 512 * 4);
  mamba2_block(st, ws, h2, 2, 512, sw.eblk_lnw, sw.eblk_lnb, sw.eblk, dm, etok);
  const int Ltot = 2560;
  float* xcat = (float*)ws.alloc((size_t)2 * Ltot * 512 * 4);
  k_copy_seq<<<gblk(2LL * 512 * 512), 256, 0, st>>>(etok, xcat, 2, 512, 512,
      512LL * 512, (long long)Ltot * 512);
  k_copy_seq<<<gblk(2LL * 1024 * 512), 256, 0, st>>>(lng, xcat + (size_t)512 * 512,
      2, 1024, 512, 1024LL * 512, (long long)Ltot * 512);
  k_copy_seq<<<gblk(2LL * 1024 * 512), 256, 0, st>>>(lnl, xcat + (size_t)1536 * 512,
      2, 1024, 512, 1024LL * 512, (long long)Ltot * 512);
  float* xsf = (float*)ws.alloc((size_t)2 * Ltot * 512 * 4);
  mamba2_block(st, ws, xcat, 2, Ltot, sw.sblk_lnw, sw.sblk_lnb, sw.sblk, dm, xsf);
  float* xs1 = (float*)ws.alloc((size_t)rowsL * 512 * 4);
  k_slice_add<<<gblk(rowsL * 512), 256, 0, st>>>(xsf, Ltot, 1536, lnl, xs1, 2, 1024, 512);
  float* xs = (float*)ws.alloc((size_t)rowsL * 512 * 4);
  k_ln<<<(unsigned)rowsL, 256, 0, st>>>(xs1, sw.l3w, sw.l3b, xs, 512);
  float* f1 = (float*)ws.alloc((size_t)rowsL * 256 * 4);
  __bf16* w1 = (__bf16*)ws.alloc((size_t)512 * 256 * 2);
  packWT(st, sw.f1w, w1, 512, 256);
  gemm(st, xs, w1, f1, (int)rowsL, 256, 512, 512, 512, 256, 0, 0, 0, 1,
       sw.f1b, nullptr, 0, 3, 0, 0, 0, 0);
  __bf16* w2 = (__bf16*)ws.alloc((size_t)256 * 512 * 2);
  packWT(st, sw.f2w, w2, 256, 512);
  gemm(st, f1, w2, out, (int)rowsL, 512, 256, 256, 256, 512, 0, 0, 0, 1,
       sw.f2b, xs, 512, 5, 0, 0, 0, 0);
  ws.used = mark;
}

// --------- input pointer mapping (setup_inputs dict insertion order) --------
struct P {
  void* const* in; int i;
  const float* nf() { return (const float*)in[i++]; }
};
static MambaW take_mamba(P& p) {
  MambaW w;
  w.W_in = p.nf(); w.conv_w = p.nf(); w.conv_b = p.nf(); w.dt_bias = p.nf();
  w.A_log = p.nf(); w.D = p.nf(); w.norm_w = p.nf(); w.W_out = p.nf();
  return w;
}
static ConvW take_conv(P& p) {
  ConvW c;
  for (int i = 0; i < 7; ++i) { c.w[i] = p.nf(); c.b[i] = p.nf(); }
  c.g1 = p.nf(); c.b1 = p.nf(); c.g2 = p.nf(); c.b2 = p.nf();
  return c;
}
static BimW take_bim(P& p) {
  BimW b;
  b.l1w = p.nf(); b.l1b = p.nf(); b.l2w = p.nf(); b.l2b = p.nf();
  b.fw = take_mamba(p); b.bw = take_mamba(p);
  b.f1w = p.nf(); b.f1b = p.nf(); b.f2w = p.nf(); b.f2b = p.nf();
  return b;
}
static StackW take_stack(P& p) {
  StackW s;
  s.l1w = p.nf(); s.l1b = p.nf(); s.l2w = p.nf(); s.l2b = p.nf();
  s.l3w = p.nf(); s.l3b = p.nf();
  s.ew = p.nf(); s.elw = p.nf(); s.elb = p.nf();
  s.eblk_lnw = p.nf(); s.eblk_lnb = p.nf(); s.eblk = take_mamba(p);
  s.sblk_lnw = p.nf(); s.sblk_lnb = p.nf(); s.sblk = take_mamba(p);
  s.f1w = p.nf(); s.f1b = p.nf(); s.f2w = p.nf(); s.f2b = p.nf();
  return s;
}

extern "C" void kernel_launch(void* const* d_in, const int* in_sizes, int n_in,
                              void* d_out, int out_size, void* d_ws, size_t ws_size,
                              hipStream_t stream) {
  (void)in_sizes; (void)n_in; (void)out_size;
  Bump ws{(char*)d_ws, 0, ws_size};
  P p{d_in, 0};
  const float* X  = p.nf();   // x        [2,1024,512]
  const float* Xe = p.nf();   // x_extra  [2,512]
  BimW  bm  = take_bim(p);
  ConvW cb1 = take_conv(p);
  ConvW cb2 = take_conv(p);
  ConvW cb3 = take_conv(p);
  ConvW cb4 = take_conv(p);
  const float* dw = p.nf();   // dense_w
  const float* db = p.nf();   // dense_b
  StackW sw = take_stack(p);

  const long long rows = 2048;
  float* xa0 = (float*)ws.alloc((size_t)rows * 512 * 4);
  run_bimamba(stream, ws, X, xa0, bm);

  float* t1 = (float*)ws.alloc((size_t)rows * 512 * 4);
  run_conv_block(stream, ws, xa0, t1, cb1, rows);

  float* xskip = (float*)ws.alloc((size_t)rows * 512 * 4);
  run_conv_block(stream, ws, t1, xskip, cb2, rows);

  float* td = (float*)ws.alloc((size_t)rows * 512 * 4);
  {
    size_t mark = ws.used;
    __bf16* wp = (__bf16*)ws.alloc((size_t)512 * 512 * 2);
    packWT(stream, dw, wp, 512, 512);
    gemm(stream, t1, wp, td, (int)rows, 512, 512, 512, 512, 512, 0, 0, 0, 1,
         db, nullptr, 0, 3, 0, 0, 0, 0);
    ws.used = mark;
  }

  float* t3 = (float*)ws.alloc((size_t)rows * 512 * 4);
  run_conv_block(stream, ws, td, t3, cb3, rows);

  float* tst = (float*)ws.alloc((size_t)rows * 512 * 4);
  run_stack(stream, ws, t3, xa0, Xe, tst, sw);

  float* t4 = (float*)ws.alloc((size_t)rows * 512 * 4);
  run_conv_block(stream, ws, tst, t4, cb4, rows);

  float* OUT = (float*)d_out;   // [2,1024,1024] = concat(xa_skip, xa)
  k_copy_cols<<<gblk(rows * 512), 256, 0, stream>>>(xskip, OUT, rows, 512, 1024, 0);
  k_copy_cols<<<gblk(rows * 512), 256, 0, stream>>>(t4,    OUT, rows, 512, 1024, 512);
}
